// LLaMA_80882824118320
// MI455X (gfx1250) — matrix operand order repaired and checked by reading, not compiled, not run on hardware
//
#include <hip/hip_runtime.h>
#include <stdint.h>

// ---------------- model constants ----------------
#define VOCAB 32000
#define DMODEL 2048
#define NH 16
#define NKVH 4
#define HD 128
#define FF 5632
#define NLAYER 2
#define SEQ 2048
#define QKVW (DMODEL + 2 * NKVH * HD)   // 3072
#define EPS 1e-5f

// ---------------- WMMA types ----------------
typedef __attribute__((ext_vector_type(8)))  float        v8f;
typedef __attribute__((ext_vector_type(8)))  unsigned int v8u;
typedef __attribute__((ext_vector_type(16))) __bf16       v16bf;
typedef __attribute__((ext_vector_type(4)))  int          v4i;

__device__ __forceinline__ v8f zero8() {
    v8f z;
#pragma unroll
    for (int i = 0; i < 8; i++) z[i] = 0.0f;
    return z;
}

__device__ __forceinline__ uint16_t f2bf(float f) {
    uint32_t u = __float_as_uint(f);
    u += 0x7FFFu + ((u >> 16) & 1u);   // round-to-nearest-even
    return (uint16_t)(u >> 16);
}

// pack two fp32 -> packed bf16x2 (low = a, high = b)
__device__ __forceinline__ uint32_t pack2bf(float a, float b) {
#if __has_builtin(__builtin_amdgcn_cvt_pk_bf16_f32)
    typedef __attribute__((ext_vector_type(2))) __bf16 v2bf;
    v2bf r = __builtin_amdgcn_cvt_pk_bf16_f32(a, b);
    return __builtin_bit_cast(uint32_t, r);
#else
    return (uint32_t)f2bf(a) | ((uint32_t)f2bf(b) << 16);
#endif
}

__device__ __forceinline__ v8f wmma_bf16(const v8u& a, const v8u& b, v8f c) {
    return __builtin_amdgcn_wmma_f32_16x16x32_bf16(
        false, __builtin_bit_cast(v16bf, a),
        false, __builtin_bit_cast(v16bf, b),
        (short)0, c, false, false);
}

// ---------------- async global->LDS copy (CDNA5), with sync fallback --------
typedef __attribute__((address_space(1))) v4i* gv4p;
typedef __attribute__((address_space(3))) v4i* lv4p;

__device__ __forceinline__ void async_ld_b128(uint16_t* lds, const uint16_t* g) {
#if __has_builtin(__builtin_amdgcn_global_load_async_to_lds_b128)
    __builtin_amdgcn_global_load_async_to_lds_b128((gv4p)g, (lv4p)lds, 0, 0);
#else
    *(uint4*)lds = *(const uint4*)g;
#endif
}
__device__ __forceinline__ void async_wait_all() {
#if __has_builtin(__builtin_amdgcn_global_load_async_to_lds_b128)
#if __has_builtin(__builtin_amdgcn_s_wait_asynccnt)
    __builtin_amdgcn_s_wait_asynccnt(0);
#else
    asm volatile("s_wait_asynccnt 0x0" ::: "memory");
#endif
#endif
}

// ---------------- weight convert+transpose: W[K,N] fp32 -> Wt[N,K] bf16 -----
__global__ __launch_bounds__(256) void wtrans_k(
    const float* __restrict__ W, uint16_t* __restrict__ Wt, int K, int N)
{
    __shared__ float t[64][65];
    const int k0 = blockIdx.x * 64;
    const int n0 = blockIdx.y * 64;
#pragma unroll
    for (int i = 0; i < 16; i++) {
        int idx = threadIdx.x + i * 256;   // 0..4095
        int kr = idx >> 6, nc = idx & 63;
        t[kr][nc] = W[(size_t)(k0 + kr) * N + n0 + nc];
    }
    __syncthreads();
#pragma unroll
    for (int i = 0; i < 16; i++) {
        int idx = threadIdx.x + i * 256;
        int nr = idx >> 6, kc = idx & 63;
        Wt[(size_t)(n0 + nr) * K + k0 + kc] = f2bf(t[kc][nr]);
    }
}

// ---------------- GEMM: C[M,N] fp32 = Ah[M,K] bf16 @ Wt[N,K]^T bf16 (+res) --
// Block tile 64(M) x 128(N), K-step 32, 128 threads (4 waves, each 64x32).
// Both operands staged into LDS by async B128 copies (no conversion in loop).
#define BM 64
#define BN 128
#define BKK 32
#define LDT 40   // BKK + 8 pad (bf16 elements); row stride 80B (16B aligned)

template <bool RES>
__global__ __launch_bounds__(128) void gemm_bf16_k(
    const uint16_t* __restrict__ Ah, const uint16_t* __restrict__ Wt,
    const float* __restrict__ res, float* __restrict__ C,
    int M, int N, int K)
{
    __shared__ uint16_t As[BM][LDT];
    __shared__ uint16_t Bs[BN][LDT];

    const int tid  = threadIdx.x;
    const int lane = tid & 31;
    const int wave = tid >> 5;
    const int wn   = wave * 32;            // wave's n-offset within BN
    const int row0 = blockIdx.y * BM;
    const int col0 = blockIdx.x * BN;
    const int m    = lane & 15;
    const int half = lane >> 4;

    v8f acc[4][2];
#pragma unroll
    for (int i = 0; i < 4; i++)
#pragma unroll
        for (int j = 0; j < 2; j++) acc[i][j] = zero8();

    for (int kk = 0; kk < K; kk += BKK) {
        // A tile: 64 rows x 32 bf16 = 256 B128 transfers (2 per thread)
#pragma unroll
        for (int i = 0; i < 2; i++) {
            int idx = tid + i * 128;
            int r = idx >> 2, c = idx & 3;
            async_ld_b128(&As[r][c * 8], Ah + (size_t)(row0 + r) * K + kk + c * 8);
        }
        // B tile from Wt[N,K]: 128 rows x 32 bf16 = 512 B128 transfers (4 per thread)
#pragma unroll
        for (int i = 0; i < 4; i++) {
            int idx = tid + i * 128;
            int n = idx >> 2, c = idx & 3;
            async_ld_b128(&Bs[n][c * 8], Wt + (size_t)(col0 + n) * K + kk + c * 8);
        }
        async_wait_all();
        __syncthreads();

        v8u a[4], b[2];
#pragma unroll
        for (int t = 0; t < 4; t++) {
            const int mm = t * 16 + m;
#pragma unroll
            for (int j = 0; j < 8; j++) {
                int kq = ((j >> 2) << 4) + (half << 3) + ((j & 3) << 1);
                a[t][j] = *(const uint32_t*)&As[mm][kq];
            }
        }
#pragma unroll
        for (int t = 0; t < 2; t++) {
            const int nn = wn + t * 16 + m;
#pragma unroll
            for (int j = 0; j < 8; j++) {
                int kq = ((j >> 2) << 4) + (half << 3) + ((j & 3) << 1);
                b[t][j] = *(const uint32_t*)&Bs[nn][kq];
            }
        }
#pragma unroll
        for (int i = 0; i < 4; i++)
#pragma unroll
            for (int j = 0; j < 2; j++) acc[i][j] = wmma_bf16(a[i], b[j], acc[i][j]);
        __syncthreads();
    }

#pragma unroll
    for (int i = 0; i < 4; i++) {
        const int gm0 = row0 + i * 16 + (half << 3);
#pragma unroll
        for (int j = 0; j < 2; j++) {
            const int gn = col0 + wn + j * 16 + m;
            const size_t base = (size_t)gm0 * N + gn;
#pragma unroll
            for (int r = 0; r < 8; r++) {
                const size_t off = base + (size_t)r * N;
                float v = acc[i][j][r];
                if (RES) v += res[off];
                C[off] = v;
            }
        }
    }
}

// ---------------- flash attention: one wave per (16-query tile, head) -------
// Reads fp32 QKV (post-RoPE), writes bf16 attention output (GEMM A operand).
__global__ __launch_bounds__(32) void flash_attn_k(
    const float* __restrict__ qkv, uint16_t* __restrict__ out)
{
    __shared__ uint16_t Ks[32][136];   // [key][d] bf16
    __shared__ uint16_t Vt[128][40];   // [d][key] bf16
    __shared__ uint16_t Ps[16][40];    // [query][key] bf16

    const int qt   = blockIdx.x;       // 0..SEQ/16-1
    const int h    = blockIdx.y;       // 0..NH-1
    const int kvh  = h >> 2;           // GQA: rep = 4
    const int lane = threadIdx.x;
    const int m    = lane & 15;
    const int half = lane >> 4;
    const int strd = QKVW;
    const float scale = 0.08838834764831845f;   // 1/sqrt(128)

    // Q fragments (A layout), pre-scaled
    v8u qa[4];
    {
        const float* qrow = qkv + (size_t)(qt * 16 + m) * strd + h * HD;
#pragma unroll
        for (int c = 0; c < 4; c++)
#pragma unroll
            for (int j = 0; j < 8; j++) {
                int d = c * 32 + ((j >> 2) << 4) + (half << 3) + ((j & 3) << 1);
                qa[c][j] = pack2bf(qrow[d] * scale, qrow[d + 1] * scale);
            }
    }

    v8f o[4];
#pragma unroll
    for (int t = 0; t < 4; t++) o[t] = zero8();
    float rmax[8], rsum[8];
#pragma unroll
    for (int r = 0; r < 8; r++) { rmax[r] = -1e30f; rsum[r] = 0.0f; }

    const int nkb = (qt * 16 + 15) / 32 + 1;
    const float* kbase = qkv + DMODEL + kvh * HD;
    const float* vbase = qkv + DMODEL + NKVH * HD + kvh * HD;

    for (int kb = 0; kb < nkb; kb++) {
        __syncthreads();
        const int key = kb * 32 + lane;
        {   // stage K row (bf16 row-major) and V row (bf16 transposed)
            const float* kr = kbase + (size_t)key * strd;
#pragma unroll
            for (int d4 = 0; d4 < 32; d4++) {
                float4 f = *(const float4*)(kr + d4 * 4);
                *(uint32_t*)&Ks[lane][d4 * 4]     = pack2bf(f.x, f.y);
                *(uint32_t*)&Ks[lane][d4 * 4 + 2] = pack2bf(f.z, f.w);
            }
            const float* vr = vbase + (size_t)key * strd;
#pragma unroll
            for (int d = 0; d < 128; d += 2) {
                uint32_t u = pack2bf(vr[d], vr[d + 1]);
                Vt[d][lane]     = (uint16_t)u;
                Vt[d + 1][lane] = (uint16_t)(u >> 16);
            }
        }
        __syncthreads();

        // scores: S[16 x 32] = Q @ K^T  (two 16x16 C fragments)
        v8f s0 = zero8(), s1 = zero8();
#pragma unroll
        for (int c = 0; c < 4; c++) {
            v8u b0, b1;
#pragma unroll
            for (int j = 0; j < 8; j++) {
                int d = c * 32 + ((j >> 2) << 4) + (half << 3) + ((j & 3) << 1);
                b0[j] = *(const uint32_t*)&Ks[m][d];
                b1[j] = *(const uint32_t*)&Ks[m + 16][d];
            }
            s0 = wmma_bf16(qa[c], b0, s0);
            s1 = wmma_bf16(qa[c], b1, s1);
        }

        // causal mask + online softmax
        float alpha[8];
#pragma unroll
        for (int r = 0; r < 8; r++) {
            const int qidx = qt * 16 + r + (half << 3);
            const int key0 = kb * 32 + m;
            if (key0 > qidx)      s0[r] = -1e30f;
            if (key0 + 16 > qidx) s1[r] = -1e30f;

            float mx = fmaxf(s0[r], s1[r]);
#pragma unroll
            for (int off = 8; off; off >>= 1) mx = fmaxf(mx, __shfl_xor(mx, off, 32));
            float mnew = fmaxf(rmax[r], mx);
            float a  = __expf(rmax[r] - mnew);
            float p0 = __expf(s0[r] - mnew);
            float p1 = __expf(s1[r] - mnew);
            s0[r] = p0; s1[r] = p1;
            float ps = p0 + p1;
#pragma unroll
            for (int off = 8; off; off >>= 1) ps += __shfl_xor(ps, off, 32);
            rsum[r] = rsum[r] * a + ps;
            rmax[r] = mnew;
            alpha[r] = a;
        }
#pragma unroll
        for (int t = 0; t < 4; t++)
#pragma unroll
            for (int r = 0; r < 8; r++) o[t][r] *= alpha[r];

        // P (C layout) -> LDS -> A layout
        __syncthreads();
#pragma unroll
        for (int r = 0; r < 8; r++) {
            const int rm = r + (half << 3);
            uint32_t u = pack2bf(s0[r], s1[r]);
            Ps[rm][m]      = (uint16_t)u;
            Ps[rm][16 + m] = (uint16_t)(u >> 16);
        }
        __syncthreads();
        v8u pa;
#pragma unroll
        for (int j = 0; j < 8; j++) {
            int kq = ((j >> 2) << 4) + (half << 3) + ((j & 3) << 1);
            pa[j] = *(const uint32_t*)&Ps[m][kq];
        }
        // O += P @ V
#pragma unroll
        for (int t = 0; t < 4; t++) {
            v8u vb;
#pragma unroll
            for (int j = 0; j < 8; j++) {
                int kq = ((j >> 2) << 4) + (half << 3) + ((j & 3) << 1);
                vb[j] = *(const uint32_t*)&Vt[t * 16 + m][kq];
            }
            o[t] = wmma_bf16(pa, vb, o[t]);
        }
    }

#pragma unroll
    for (int t = 0; t < 4; t++)
#pragma unroll
        for (int r = 0; r < 8; r++) {
            const int qidx = qt * 16 + r + (half << 3);
            out[(size_t)qidx * DMODEL + h * HD + t * 16 + m] = f2bf(o[t][r] / rsum[r]);
        }
}

// ---------------- small elementwise kernels ----------------
// RMSNorm: fp32 in -> bf16 out (feeds GEMM A operand)
__global__ __launch_bounds__(256) void rmsnorm_k(
    const float* __restrict__ x, const float* __restrict__ w, uint16_t* __restrict__ out)
{
    __shared__ float red[256];
    const int row = blockIdx.x;
    const float* xr = x + (size_t)row * DMODEL;
    float ss = 0.0f;
    for (int i = threadIdx.x; i < DMODEL; i += 256) { float v = xr[i]; ss += v * v; }
    red[threadIdx.x] = ss;
    __syncthreads();
    for (int s = 128; s > 0; s >>= 1) {
        if (threadIdx.x < s) red[threadIdx.x] += red[threadIdx.x + s];
        __syncthreads();
    }
    const float inv = rsqrtf(red[0] / (float)DMODEL + EPS);
    for (int i = threadIdx.x; i < DMODEL; i += 256)
        out[(size_t)row * DMODEL + i] = f2bf(xr[i] * inv * w[i]);
}

__global__ __launch_bounds__(256) void rope_k(
    float* __restrict__ qkv, const float* __restrict__ fc)
{
    const int idx  = blockIdx.x * 256 + threadIdx.x;   // pair index
    const int i    = idx & 63;                         // pair within head
    const int head = (idx >> 6) % (NH + NKVH);
    const int s    = idx / (64 * (NH + NKVH));
    const int col  = (head < NH) ? head * HD + 2 * i
                                 : DMODEL + (head - NH) * HD + 2 * i;
    float* p = qkv + (size_t)s * QKVW + col;
    const float c  = fc[((size_t)s * 64 + i) * 2 + 0];
    const float sn = fc[((size_t)s * 64 + i) * 2 + 1];
    const float x0 = p[0], x1 = p[1];
    p[0] = x0 * c - x1 * sn;
    p[1] = x1 * c + x0 * sn;
}

__global__ __launch_bounds__(256) void embed_k(
    const int* __restrict__ ids, const float* __restrict__ wemb, float* __restrict__ out)
{
    const int idx = blockIdx.x * 256 + threadIdx.x;    // S*D elements
    const int s = idx >> 11;
    const int d = idx & (DMODEL - 1);
    out[idx] = wemb[(size_t)ids[s] * DMODEL + d];
}

// SwiGLU: g1h = bf16( silu(g1) * g3 )
__global__ __launch_bounds__(256) void silu_mul_k(
    const float* __restrict__ g1, const float* __restrict__ g3,
    uint16_t* __restrict__ g1h, int n)
{
    const int i = blockIdx.x * 256 + threadIdx.x;
    if (i < n) {
        const float x = g1[i];
        g1h[i] = f2bf((x / (1.0f + __expf(-x))) * g3[i]);
    }
}

// ---------------- launch ----------------
extern "C" void kernel_launch(void* const* d_in, const int* in_sizes, int n_in,
                              void* d_out, int out_size, void* d_ws, size_t ws_size,
                              hipStream_t stream)
{
    (void)in_sizes; (void)n_in; (void)out_size; (void)ws_size;
    const int*   ids    = (const int*)d_in[0];
    // d_in[1] = mask (bool), unused: causal mask computed analytically
    const float* fc     = (const float*)d_in[2];
    const float* wemb   = (const float*)d_in[3];
    const float* wqkv   = (const float*)d_in[4];
    const float* wo     = (const float*)d_in[5];
    const float* w1     = (const float*)d_in[6];
    const float* w3     = (const float*)d_in[7];
    const float* w2     = (const float*)d_in[8];
    const float* anorm  = (const float*)d_in[9];
    const float* fnorm  = (const float*)d_in[10];
    const float* finaln = (const float*)d_in[11];
    const float* whead  = (const float*)d_in[12];

    // ---- workspace carve (fp32 region, then bf16 region) ----
    float* wsf = (float*)d_ws;
    size_t off = 0;
    float* X0  = wsf + off; off += (size_t)SEQ * DMODEL;
    float* X1  = wsf + off; off += (size_t)SEQ * DMODEL;
    float* QKV = wsf + off; off += (size_t)SEQ * QKVW;
    float* G1  = wsf + off; off += (size_t)SEQ * FF;
    float* G3  = wsf + off; off += (size_t)SEQ * FF;
    uint16_t* wsh = (uint16_t*)(wsf + off);
    size_t hoff = 0;
    uint16_t* XNh = wsh + hoff; hoff += (size_t)SEQ * DMODEL;   // normed x (bf16)
    uint16_t* AOh = wsh + hoff; hoff += (size_t)SEQ * DMODEL;   // attn out (bf16)
    uint16_t* G1h = wsh + hoff; hoff += (size_t)SEQ * FF;       // swiglu out (bf16)
    uint16_t* WT  = wsh + hoff; hoff += (size_t)VOCAB * DMODEL; // Wt scratch (max N*K)

    embed_k<<<(SEQ * DMODEL) / 256, 256, 0, stream>>>(ids, wemb, X0);

    for (int l = 0; l < NLAYER; l++) {
        rmsnorm_k<<<SEQ, 256, 0, stream>>>(X0, anorm + (size_t)l * DMODEL, XNh);
        wtrans_k<<<dim3(DMODEL / 64, QKVW / 64), 256, 0, stream>>>(
            wqkv + (size_t)l * DMODEL * QKVW, WT, DMODEL, QKVW);
        gemm_bf16_k<false><<<dim3(QKVW / BN, SEQ / BM), 128, 0, stream>>>(
            XNh, WT, nullptr, QKV, SEQ, QKVW, DMODEL);
        rope_k<<<(SEQ * (NH + NKVH) * 64) / 256, 256, 0, stream>>>(QKV, fc);
        flash_attn_k<<<dim3(SEQ / 16, NH), 32, 0, stream>>>(QKV, AOh);
        wtrans_k<<<dim3(DMODEL / 64, DMODEL / 64), 256, 0, stream>>>(
            wo + (size_t)l * DMODEL * DMODEL, WT, DMODEL, DMODEL);
        gemm_bf16_k<true><<<dim3(DMODEL / BN, SEQ / BM), 128, 0, stream>>>(
            AOh, WT, X0, X1, SEQ, DMODEL, DMODEL);
        rmsnorm_k<<<SEQ, 256, 0, stream>>>(X1, fnorm + (size_t)l * DMODEL, XNh);
        wtrans_k<<<dim3(DMODEL / 64, FF / 64), 256, 0, stream>>>(
            w1 + (size_t)l * DMODEL * FF, WT, DMODEL, FF);
        gemm_bf16_k<false><<<dim3(FF / BN, SEQ / BM), 128, 0, stream>>>(
            XNh, WT, nullptr, G1, SEQ, FF, DMODEL);
        wtrans_k<<<dim3(DMODEL / 64, FF / 64), 256, 0, stream>>>(
            w3 + (size_t)l * DMODEL * FF, WT, DMODEL, FF);
        gemm_bf16_k<false><<<dim3(FF / BN, SEQ / BM), 128, 0, stream>>>(
            XNh, WT, nullptr, G3, SEQ, FF, DMODEL);
        silu_mul_k<<<(SEQ * FF) / 256, 256, 0, stream>>>(G1, G3, G1h, SEQ * FF);
        wtrans_k<<<dim3(FF / 64, DMODEL / 64), 256, 0, stream>>>(
            w2 + (size_t)l * FF * DMODEL, WT, FF, DMODEL);
        gemm_bf16_k<true><<<dim3(DMODEL / BN, SEQ / BM), 128, 0, stream>>>(
            G1h, WT, X1, X0, SEQ, DMODEL, FF);
    }

    rmsnorm_k<<<SEQ, 256, 0, stream>>>(X0, finaln, XNh);
    wtrans_k<<<dim3(DMODEL / 64, VOCAB / 64), 256, 0, stream>>>(
        whead, WT, DMODEL, VOCAB);
    gemm_bf16_k<false><<<dim3(VOCAB / BN, SEQ / BM), 128, 0, stream>>>(
        XNh, WT, nullptr, (float*)d_out, SEQ, VOCAB, DMODEL);
}
